// SelfAttention_57312043597833
// MI455X (gfx1250) — matrix operand order repaired and checked by reading, not compiled, not run on hardware
//
#include <hip/hip_runtime.h>

// ---------------------------------------------------------------------------
// CDNA5 (gfx1250, wave32) GQA self-attention.
//  * All matmuls: v_wmma_f32_16x16x32_f16 (D = A(16x32 f16) x B(32x16 f16) + C f32)
//  * GEMM tiles staged to LDS by the Tensor Data Mover (tensor_load_to_lds,
//    TENSORcnt-tracked, double-buffered) when the builtin is available;
//    synchronous global->LDS staging otherwise.
//
// Fragment reads (CDNA5 ISA 7.12.2, wave32) all fall inside one matrix
// row, so row-major LDS tiles suffice:
//   A (f16 16x32): lane = half*16+m; elem i -> k=half*16+i  => 32B at row off half*16
//   B (f16 32x16): lane = half*16+n; elem i -> k=half*16+i  => 32B at row off half*16
//   C/D (f32 16x16): lane holds column n=lane&15; VGPR v holds row m=v+8*half
// ---------------------------------------------------------------------------

typedef __attribute__((ext_vector_type(16))) _Float16     v16h;
typedef __attribute__((ext_vector_type(8)))  float        v8f;
typedef __attribute__((ext_vector_type(4)))  unsigned int v4u;
typedef __attribute__((ext_vector_type(8)))  int          v8i;
typedef __attribute__((ext_vector_type(4)))  int          v4i;

#if defined(__has_builtin)
#if __has_builtin(__builtin_amdgcn_tensor_load_to_lds)
#define HAVE_TDM 1
#endif
#endif
#ifndef HAVE_TDM
#define HAVE_TDM 0
#endif

union Frag16 {
    v16h   v;
    float4 f4[2];
};

__device__ inline v8f zero_v8f() {
    v8f z;
#pragma unroll
    for (int i = 0; i < 8; ++i) z[i] = 0.0f;
    return z;
}

__device__ inline void wait_tensorcnt0() {
#if defined(__has_builtin)
#if __has_builtin(__builtin_amdgcn_s_wait_tensorcnt)
    __builtin_amdgcn_s_wait_tensorcnt(0);
#else
    asm volatile("s_wait_tensorcnt 0" ::: "memory");
#endif
#else
    asm volatile("s_wait_tensorcnt 0" ::: "memory");
#endif
}

#if HAVE_TDM
// Issue a TDM 2D tile load: rows x row_elems f16 tile from a row-major tensor
// (row stride = row_stride elems) into LDS at lds_off (row-major, packed).
__device__ inline void tdm_load_tile(const _Float16* gsrc, unsigned lds_off,
                                     unsigned rows, unsigned row_elems,
                                     unsigned tensor_rows, unsigned tensor_cols,
                                     unsigned long long row_stride) {
    unsigned long long ga = (unsigned long long)(size_t)gsrc;
    v4u g0;
    g0[0] = 1u;                                   // count=1, user mode, no gather
    g0[1] = lds_off;                              // D#.lds_addr (bytes)
    g0[2] = (unsigned)(ga & 0xFFFFFFFFull);       // global_addr[31:0]
    g0[3] = (unsigned)((ga >> 32) & 0x01FFFFFFull) | 0x80000000u;  // [56:32] | type=2
    v8i g1;
    g1[0] = (int)(1u << 16);                      // workgroup_mask=0, data_size=1 (2B)
    g1[1] = (int)((tensor_cols & 0xFFFFu) << 16); // tensor_dim0[15:0] @ bits 63:48
    g1[2] = (int)((tensor_cols >> 16) | ((tensor_rows & 0xFFFFu) << 16));
    g1[3] = (int)((tensor_rows >> 16) | ((row_elems & 0xFFFFu) << 16)); // tile_dim0
    g1[4] = (int)(rows & 0xFFFFu);                // tile_dim1 (tile_dim2=0)
    g1[5] = (int)(unsigned)(row_stride & 0xFFFFFFFFull);   // tensor_dim0_stride lo
    g1[6] = (int)(unsigned)((row_stride >> 32) & 0xFFFFull);
    g1[7] = 0;                                    // tensor_dim1_stride (unused, 2D)
    v4i gz;
#pragma unroll
    for (int i = 0; i < 4; ++i) gz[i] = 0;
#if __clang_major__ >= 23
    v8i gz8;
#pragma unroll
    for (int i = 0; i < 8; ++i) gz8[i] = 0;
    __builtin_amdgcn_tensor_load_to_lds(g0, g1, gz, gz, gz8, 0);
#else
    __builtin_amdgcn_tensor_load_to_lds(g0, g1, gz, gz, 0);
#endif
}
#endif  // HAVE_TDM

// ---------------------------------------------------------------------------
// Elementwise / transpose converters
// ---------------------------------------------------------------------------
__global__ __launch_bounds__(256) void cvt_f32_to_f16(const float* __restrict__ in,
                                                      _Float16* __restrict__ out,
                                                      size_t n) {
    size_t i = (size_t)blockIdx.x * 256 + threadIdx.x;
    if (i < n) out[i] = (_Float16)in[i];
}

// out[c][r] = (f16) in[r][c];  in is [R][C] row-major
__global__ __launch_bounds__(256) void transpose_f32_to_f16(const float* __restrict__ in,
                                                            _Float16* __restrict__ out,
                                                            int R, int C) {
    size_t i = (size_t)blockIdx.x * 256 + threadIdx.x;
    if (i >= (size_t)R * C) return;
    int c = (int)(i % C);
    int r = (int)(i / C);
    out[(size_t)c * R + r] = (_Float16)in[i];
}

// ---------------------------------------------------------------------------
// Shared WMMA compute step on one (128x32 A-tile, 64x32 B-tile) LDS pair.
// lA: row-major 128x32 f16, lB: row-major 64x32 f16 (rows = Bt rows, i.e. N).
// ---------------------------------------------------------------------------
__device__ inline void gemm_compute_step(const _Float16* lA, const _Float16* lB,
                                         int wave, int half, int nl, v8f acc[4]) {
    Frag16 a;
    const _Float16* ar = lA + (wave * 16 + nl) * 32;
    a.f4[0] = *reinterpret_cast<const float4*>(ar + half * 16);
    a.f4[1] = *reinterpret_cast<const float4*>(ar + half * 16 + 8);
    Frag16 b[4];
#pragma unroll
    for (int nt = 0; nt < 4; ++nt) {   // issue all B-frag loads first
        const _Float16* br = lB + (nt * 16 + nl) * 32 + half * 16;
        b[nt].f4[0] = *reinterpret_cast<const float4*>(br);
        b[nt].f4[1] = *reinterpret_cast<const float4*>(br + 8);
    }
#pragma unroll
    for (int nt = 0; nt < 4; ++nt) {   // then WMMAs back-to-back
        acc[nt] = __builtin_amdgcn_wmma_f32_16x16x32_f16(
            false, a.v, false, b[nt].v, (short)0, acc[nt], false, false);
    }
}

// ---------------------------------------------------------------------------
// GEMM: C[M][N] = A[M][K] * B[K][N]; A f16 row-major, Bt = B^T ([N][K]) f16,
// C f32.  Tile 128(M) x 64(N), K-step 32, 256 threads = 8 waves.
// TDM path: double-buffered LDS, wave 0 drives the Tensor Data Mover.
// ---------------------------------------------------------------------------
__global__ __launch_bounds__(256) void gemm_f16_wmma(const _Float16* __restrict__ A,
                                                     const _Float16* __restrict__ Bt,
                                                     float* __restrict__ C,
                                                     int M, int N, int K) {
    const int tid  = threadIdx.x;
    const int lane = tid & 31;
    const int wave = tid >> 5;
    const int half = lane >> 4;
    const int nl   = lane & 15;

    const int m0 = blockIdx.x * 128;
    const int n0 = blockIdx.y * 64;

    v8f acc[4];
#pragma unroll
    for (int i = 0; i < 4; ++i) acc[i] = zero_v8f();

#if HAVE_TDM
    __shared__ __align__(16) _Float16 lsA[2][128 * 32];
    __shared__ __align__(16) _Float16 lsB[2][64 * 32];

    const int nk = K / 32;
    if (wave == 0) {
        tdm_load_tile(A + (size_t)m0 * K, (unsigned)(size_t)&lsA[0][0],
                      128, 32, (unsigned)M, (unsigned)K, (unsigned long long)K);
        tdm_load_tile(Bt + (size_t)n0 * K, (unsigned)(size_t)&lsB[0][0],
                      64, 32, (unsigned)N, (unsigned)K, (unsigned long long)K);
        wait_tensorcnt0();
    }
    __syncthreads();
    for (int kt = 0; kt < nk; ++kt) {
        const int cur = kt & 1;
        if (kt + 1 < nk && wave == 0) {   // DMA next tiles while we compute
            const int k1 = (kt + 1) * 32;
            tdm_load_tile(A + (size_t)m0 * K + k1, (unsigned)(size_t)&lsA[cur ^ 1][0],
                          128, 32, (unsigned)M, (unsigned)K, (unsigned long long)K);
            tdm_load_tile(Bt + (size_t)n0 * K + k1, (unsigned)(size_t)&lsB[cur ^ 1][0],
                          64, 32, (unsigned)N, (unsigned)K, (unsigned long long)K);
        }
        gemm_compute_step(&lsA[cur][0], &lsB[cur][0], wave, half, nl, acc);
        if (kt + 1 < nk) {
            if (wave == 0) wait_tensorcnt0();
            __syncthreads();
        }
    }
#else
    __shared__ __align__(16) _Float16 lsA[128 * 32];
    __shared__ __align__(16) _Float16 lsB[64 * 32];

    for (int k0 = 0; k0 < K; k0 += 32) {
        __syncthreads();
#pragma unroll
        for (int cc = 0; cc < 2; ++cc) {           // A: 2x16B chunks per thread
            int c = tid * 2 + cc, row = c >> 2, jj = c & 3;
            *reinterpret_cast<float4*>(lsA + row * 32 + jj * 8) =
                *reinterpret_cast<const float4*>(A + (size_t)(m0 + row) * K + k0 + jj * 8);
        }
        {                                          // B: 1x16B chunk per thread
            int n = tid >> 2, jj = tid & 3;
            *reinterpret_cast<float4*>(lsB + n * 32 + jj * 8) =
                *reinterpret_cast<const float4*>(Bt + (size_t)(n0 + n) * K + k0 + jj * 8);
        }
        __syncthreads();
        gemm_compute_step(lsA, lsB, wave, half, nl, acc);
    }
#endif

    const int mrow = m0 + wave * 16 + 8 * half;
#pragma unroll
    for (int nt = 0; nt < 4; ++nt) {
        int ncol = n0 + nt * 16 + nl;
#pragma unroll
        for (int v = 0; v < 8; ++v)
            C[(size_t)(mrow + v) * N + ncol] = acc[nt][v];
    }
}

// ---------------------------------------------------------------------------
// RoPE + reorg.  qkv f32 is [B][T][1536] with f = ((g*6 + j)*64 + d),
// j<4 -> q head h=g*4+j, j==4 -> k, j==5 -> v.  RoPE on first 32 dims of q,k.
// Outputs (f16): q [B][16][T][64], k [B][4][T][64], vT [B][4][64][T].
// ---------------------------------------------------------------------------
__global__ __launch_bounds__(256) void rope_reorg(const float* __restrict__ qkv,
                                                  const float* __restrict__ cosb,
                                                  const float* __restrict__ sinb,
                                                  _Float16* __restrict__ qb,
                                                  _Float16* __restrict__ kb,
                                                  _Float16* __restrict__ vT,
                                                  int T, size_t total) {
    size_t idx = (size_t)blockIdx.x * 256 + threadIdx.x;
    if (idx >= total) return;
    int    f  = (int)(idx % 1536);
    size_t bt = idx / 1536;
    int    t  = (int)(bt % T);
    int    b  = (int)(bt / T);
    int g = f / 384, r = f % 384, j = r / 64, d = r % 64;

    float val = qkv[idx];
    if (j < 5 && d < 32) {  // rope on q (j<4) and k (j==4), dims 0..31
        float cs = cosb[(size_t)t * 32 + d];
        float sn = sinb[(size_t)t * 32 + d];
        float partner = qkv[idx + (d < 16 ? 16 : -16)];
        float rot = (d < 16) ? -partner : partner;
        val = val * cs + rot * sn;
    }
    _Float16 hv = (_Float16)val;
    if (j < 4) {
        int h = g * 4 + j;
        qb[(((size_t)b * 16 + h) * T + t) * 64 + d] = hv;
    } else if (j == 4) {
        kb[(((size_t)b * 4 + g) * T + t) * 64 + d] = hv;
    } else {
        vT[(((size_t)b * 4 + g) * 64 + d) * (size_t)T + t] = hv;
    }
}

// ---------------------------------------------------------------------------
// Flash attention: grid (B*16, T/128); 8 waves x 16 query rows each.
// Q in registers; per 32-wide s-tile: S = Q K^T (4 WMMA), online softmax
// (shfl_xor reductions over the 16 lanes sharing a column half), P via
// per-wave LDS tile into an A-fragment, y += P V (4 WMMA from V^T rows).
// ---------------------------------------------------------------------------
__global__ __launch_bounds__(256) void attn_kernel(const _Float16* __restrict__ qb,
                                                   const _Float16* __restrict__ kb,
                                                   const _Float16* __restrict__ vT,
                                                   _Float16* __restrict__ yb,
                                                   int T) {
    __shared__ __align__(16) _Float16 lsP[8][512];  // per-wave 16x32 P tile

    const int tid  = threadIdx.x;
    const int lane = tid & 31;
    const int wave = tid >> 5;
    const int half = lane >> 4;
    const int nl   = lane & 15;

    const int bh = blockIdx.x;
    const int b  = bh >> 4;
    const int h  = bh & 15;
    const int g  = h >> 2;
    const int t0 = blockIdx.y * 128 + wave * 16;

    const _Float16* qbase = qb + (((size_t)b * 16 + h) * T + t0) * 64;
    const _Float16* kbase = kb + (((size_t)b * 4 + g) * T) * 64;
    const _Float16* vbase = vT + (((size_t)b * 4 + g) * 64) * (size_t)T;

    Frag16 qf[2];
#pragma unroll
    for (int c = 0; c < 2; ++c) {
        const _Float16* qr = qbase + nl * 64 + c * 32;
        qf[c].f4[0] = *reinterpret_cast<const float4*>(qr + half * 16);
        qf[c].f4[1] = *reinterpret_cast<const float4*>(qr + half * 16 + 8);
    }

    float mst[8], lst[8];
    v8f yacc[4];
#pragma unroll
    for (int v = 0; v < 8; ++v) { mst[v] = -1e30f; lst[v] = 0.0f; }
#pragma unroll
    for (int dt = 0; dt < 4; ++dt) yacc[dt] = zero_v8f();

    const float scale = 0.125f;  // 1/sqrt(64)

    for (int s0 = 0; s0 < T; s0 += 32) {
        // prefetch next s-tile K / V^T lines (global_prefetch_b8)
        if (s0 + 32 < T) {
            __builtin_prefetch(kbase + (size_t)(s0 + 32 + lane) * 64, 0, 0);
            __builtin_prefetch(vbase + (size_t)(lane) * T + s0 + 32, 0, 0);
        }
        // ---- scores S (16 x 32): all 4 K-fragments, then 4 WMMAs
        Frag16 kf[4];
#pragma unroll
        for (int nt = 0; nt < 2; ++nt)
#pragma unroll
            for (int c = 0; c < 2; ++c) {
                const _Float16* kr =
                    kbase + (size_t)(s0 + nt * 16 + nl) * 64 + c * 32 + half * 16;
                kf[nt * 2 + c].f4[0] = *reinterpret_cast<const float4*>(kr);
                kf[nt * 2 + c].f4[1] = *reinterpret_cast<const float4*>(kr + 8);
            }
        v8f sacc[2];
#pragma unroll
        for (int nt = 0; nt < 2; ++nt) {
            sacc[nt] = zero_v8f();
#pragma unroll
            for (int c = 0; c < 2; ++c)
                sacc[nt] = __builtin_amdgcn_wmma_f32_16x16x32_f16(
                    false, qf[c].v, false, kf[nt * 2 + c].v, (short)0, sacc[nt],
                    false, false);
        }
        // ---- online softmax per row m = v + 8*half
#pragma unroll
        for (int v = 0; v < 8; ++v) {
            float s0v = sacc[0][v] * scale;
            float s1v = sacc[1][v] * scale;
            float mx = fmaxf(s0v, s1v);
#pragma unroll
            for (int off = 1; off < 16; off <<= 1)
                mx = fmaxf(mx, __shfl_xor(mx, off, 32));
            float mnew = fmaxf(mst[v], mx);
            float resc = __expf(mst[v] - mnew);
            mst[v] = mnew;
            float p0 = __expf(s0v - mnew);
            float p1 = __expf(s1v - mnew);
            float rs = p0 + p1;
#pragma unroll
            for (int off = 1; off < 16; off <<= 1)
                rs += __shfl_xor(rs, off, 32);
            lst[v] = lst[v] * resc + rs;
#pragma unroll
            for (int dt = 0; dt < 4; ++dt) yacc[dt][v] *= resc;
            int prow = v + 8 * half;
            lsP[wave][prow * 32 + nl]      = (_Float16)p0;
            lsP[wave][prow * 32 + 16 + nl] = (_Float16)p1;
        }
        // ---- P as A-fragment (same-wave LDS round trip)
        Frag16 pf;
        {
            const _Float16* pr = &lsP[wave][nl * 32];
            pf.f4[0] = *reinterpret_cast<const float4*>(pr + half * 16);
            pf.f4[1] = *reinterpret_cast<const float4*>(pr + half * 16 + 8);
        }
        // ---- y += P * V : all 4 V-fragments, then 4 WMMAs
        Frag16 vf[4];
#pragma unroll
        for (int dt = 0; dt < 4; ++dt) {
            const _Float16* vr = vbase + (size_t)(dt * 16 + nl) * T + s0 + half * 16;
            vf[dt].f4[0] = *reinterpret_cast<const float4*>(vr);
            vf[dt].f4[1] = *reinterpret_cast<const float4*>(vr + 8);
        }
#pragma unroll
        for (int dt = 0; dt < 4; ++dt)
            yacc[dt] = __builtin_amdgcn_wmma_f32_16x16x32_f16(
                false, pf.v, false, vf[dt].v, (short)0, yacc[dt], false, false);
    }

    float invl[8];
#pragma unroll
    for (int v = 0; v < 8; ++v) invl[v] = 1.0f / lst[v];
#pragma unroll
    for (int dt = 0; dt < 4; ++dt) {
#pragma unroll
        for (int v = 0; v < 8; ++v) {
            size_t t = (size_t)t0 + v + 8 * half;
            yb[((size_t)b * T + t) * 1024 + (size_t)h * 64 + dt * 16 + nl] =
                (_Float16)(yacc[dt][v] * invl[v]);
        }
    }
}

// ---------------------------------------------------------------------------
// Host launcher
// ---------------------------------------------------------------------------
extern "C" void kernel_launch(void* const* d_in, const int* in_sizes, int n_in,
                              void* d_out, int out_size, void* d_ws, size_t ws_size,
                              hipStream_t stream) {
    (void)n_in; (void)out_size; (void)ws_size;
    const float* x    = (const float*)d_in[0];
    const float* cosb = (const float*)d_in[1];
    const float* sinb = (const float*)d_in[2];
    // d_in[3] = mask : reference ignores it
    const float* Wqkv = (const float*)d_in[4];
    const float* Wout = (const float*)d_in[5];

    const int T    = in_sizes[1] / 32;          // 2048
    const int Mtok = in_sizes[0] / 1024;        // B*T = 4096
    const int Bv   = Mtok / T;                  // 2

    char*  ws  = (char*)d_ws;
    size_t off = 0;
    auto alloc = [&](size_t bytes) -> void* {
        void* p = ws + off;
        off = (off + bytes + 255) & ~(size_t)255;
        return p;
    };

    _Float16* xb    = (_Float16*)alloc((size_t)Mtok * 1024 * 2);
    _Float16* WqkvT = (_Float16*)alloc((size_t)1536 * 1024 * 2);
    _Float16* WoutT = (_Float16*)alloc((size_t)1024 * 1024 * 2);
    float*    qkv   = (float*)   alloc((size_t)Mtok * 1536 * 4);
    _Float16* qb    = (_Float16*)alloc((size_t)Bv * 16 * T * 64 * 2);
    _Float16* kb    = (_Float16*)alloc((size_t)Bv * 4 * T * 64 * 2);
    _Float16* vT    = (_Float16*)alloc((size_t)Bv * 4 * 64 * T * 2);
    _Float16* yb    = (_Float16*)alloc((size_t)Mtok * 1024 * 2);

    {
        size_t n = (size_t)Mtok * 1024;
        cvt_f32_to_f16<<<(unsigned)((n + 255) / 256), 256, 0, stream>>>(x, xb, n);
    }
    transpose_f32_to_f16<<<(1024 * 1536 + 255) / 256, 256, 0, stream>>>(Wqkv, WqkvT, 1024, 1536);
    transpose_f32_to_f16<<<(1024 * 1024 + 255) / 256, 256, 0, stream>>>(Wout, WoutT, 1024, 1024);

    // qkv = x @ Wqkv   (4096 x 1536 x 1024)
    gemm_f16_wmma<<<dim3(Mtok / 128, 1536 / 64), 256, 0, stream>>>(
        xb, WqkvT, qkv, Mtok, 1536, 1024);

    {
        size_t n = (size_t)Mtok * 1536;
        rope_reorg<<<(unsigned)((n + 255) / 256), 256, 0, stream>>>(
            qkv, cosb, sinb, qb, kb, vT, T, n);
    }

    attn_kernel<<<dim3(Bv * 16, T / 128), 256, 0, stream>>>(qb, kb, vT, yb, T);

    // out = y @ Wout   (4096 x 1024 x 1024), f32 output
    gemm_f16_wmma<<<dim3(Mtok / 128, 1024 / 64), 256, 0, stream>>>(
        yb, WoutT, (float*)d_out, Mtok, 1024, 1024);
}
